// Codebook_27685359190170
// MI455X (gfx1250) — matrix operand order repaired and checked by reading, not compiled, not run on hardware
//
#include <hip/hip_runtime.h>
#include <hip/hip_bf16.h>

typedef __attribute__((ext_vector_type(16))) _Float16     v16h;
typedef __attribute__((ext_vector_type(8)))  float        v8f;
typedef __attribute__((ext_vector_type(4)))  unsigned int v4u;
typedef __attribute__((ext_vector_type(8)))  int          v8i;
typedef __attribute__((ext_vector_type(4)))  int          v4i;

struct B32bytes { uint4 a, b; };   // 32-byte, 16-byte-aligned carrier for v16h LDS loads

#define KCODES 4096
#define DDIM   256
#define MROWS  128        // rows per workgroup (8 waves x 16)
#define CODEBLK 32        // codes staged per TDM tile
#define BPAD   40         // padded LDS stride (halves): 32 data + 8 pad (=64B + 16B)

// ---------------------------------------------------------------------------
// Prep: embedT[d][k] = (f16)embed[k][d];  halfesq[k] = 0.5 * sum_d embed[k][d]^2
// ---------------------------------------------------------------------------
__global__ __launch_bounds__(DDIM)
void vq_prep_kernel(const float* __restrict__ embed,
                    _Float16* __restrict__ embedT,
                    float* __restrict__ halfesq)
{
    const int k = blockIdx.x;      // 0..K-1
    const int d = threadIdx.x;     // 0..D-1
    float v = embed[(size_t)k * DDIM + d];
    embedT[(size_t)d * KCODES + k] = (_Float16)v;

    float s = v * v;
    #pragma unroll
    for (int off = 16; off > 0; off >>= 1)
        s += __shfl_down(s, off, 32);

    __shared__ float part[8];
    if ((threadIdx.x & 31) == 0) part[threadIdx.x >> 5] = s;
    __syncthreads();
    if (threadIdx.x == 0) {
        float t = 0.f;
        #pragma unroll
        for (int i = 0; i < 8; i++) t += part[i];
        halfesq[k] = 0.5f * t;
    }
}

// ---------------------------------------------------------------------------
// Main: per-wave 16-row tile, WMMA f32<=f16 over all 4096 codes, running
// argmax.  B tiles double-buffered in LDS via Tensor Data Mover DMA.
// ---------------------------------------------------------------------------
__global__ __launch_bounds__(256)
void vq_assign_kernel(const float* __restrict__ x,
                      const float* __restrict__ embed,     // fp32 [K][D] for gather
                      const _Float16* __restrict__ embedT, // f16  [D][K]
                      const float* __restrict__ halfesq,   // [K]
                      float* __restrict__ quant,           // [T][D]
                      float* __restrict__ idx_out)         // [T] (as float)
{
    __shared__ _Float16 Bs[2][DDIM * BPAD];   // 2 x 20 KB double-buffered B tile
    __shared__ float    EsAll[KCODES];        // 16 KB: all 0.5*||e||^2

    const int t    = threadIdx.x;
    const int wave = t >> 5;
    const int lane = t & 31;
    const int hl   = lane >> 4;    // half-wave select
    const int nloc = lane & 15;    // column / A-row within tile

    const long rowBase = (long)blockIdx.x * MROWS + wave * 16;
    const uint64_t ebase = (uint64_t)(uintptr_t)embedT;

    // ---- TDM descriptor-based staging of one 256(d) x 32(code) f16 tile.
    // Hardware inserts 16B LDS padding after each 64B row: pad_interval=3
    // (16 DWORDs), pad_amount=3 (4 DWORDs)  ->  BPAD = 40 halves.
    auto stageTDM = [&](int buf, int c0n) {
        unsigned lds_addr = (unsigned)(uintptr_t)&Bs[buf][0];
        uint64_t ga = ebase + (uint64_t)c0n * sizeof(_Float16);
        v4u g0;
        g0.x = 1u;                                             // count=1 (user D#)
        g0.y = lds_addr;                                       // lds_addr[31:0]
        g0.z = (unsigned)ga;                                   // global_addr[31:0]
        g0.w = (unsigned)((ga >> 32) & 0x1FFFFFFu) | (2u << 30); // addr[56:32] | type=2
        v8i g1;
        g1[0] = (int)((1u << 16)     // data_size = 1 -> 2 bytes
                    | (1u << 20)     // pad_enable
                    | (3u << 22)     // pad_interval: 16 DWORDs (64B) of data
                    | (3u << 25));   // pad_amount:   4 DWORDs (16B) of pad
        g1[1] = (int)((KCODES & 0xFFFFu) << 16);               // tensor_dim0[15:0]
        g1[2] = (int)(((KCODES >> 16) & 0xFFFFu)
                    | ((DDIM & 0xFFFFu) << 16));               // tensor_dim1[15:0]
        g1[3] = (int)(((DDIM >> 16) & 0xFFFFu)
                    | ((unsigned)CODEBLK << 16));              // tile_dim0 = 32
        g1[4] = (int)(DDIM & 0xFFFFu);                         // tile_dim1 = 256
        g1[5] = (int)KCODES;                                   // tensor_dim0_stride
        g1[6] = 0;
        g1[7] = 0;
        v4i gz4 = {0, 0, 0, 0};                                // 2-D: groups 2/3 unused
        v8i gz8 = {0, 0, 0, 0, 0, 0, 0, 0};
        __builtin_amdgcn_tensor_load_to_lds(g0, g1, gz4, gz4, gz8, 0);
    };

    // ---- Preload all esq into LDS (one time), and load A fragments.
    #pragma unroll
    for (int i = 0; i < KCODES / 256; i++)
        EsAll[t + i * 256] = halfesq[t + i * 256];

    // A fragments (16x256 f16), documented 16-bit A layout:
    // lane: m = lane&15 (both halves); element e: p=e>>1, w=e&1,
    // K(e) = (p<4 ? 0 : 16) + 8*hl + 2*(p&3) + w
    v16h A[8];
    {
        const float* xrow = x + (rowBase + nloc) * DDIM;
        #pragma unroll
        for (int c = 0; c < 8; c++) {
            v16h a;
            #pragma unroll
            for (int p = 0; p < 8; p++) {
                int kk = c * 32 + ((p < 4) ? 0 : 16) + 8 * hl + 2 * (p & 3);
                float2 f = *(const float2*)(xrow + kk);
                a[2 * p]     = (_Float16)f.x;
                a[2 * p + 1] = (_Float16)f.y;
            }
            A[c] = a;
        }
    }

    float bestv[8];
    int   besti[8];
    #pragma unroll
    for (int r = 0; r < 8; r++) { bestv[r] = -3.402823466e38f; besti[r] = 0; }

    if (wave == 0) stageTDM(0, 0);         // prologue: first tile in flight

    const int NBLK = KCODES / CODEBLK;     // 128
    for (int cb = 0; cb < NBLK; cb++) {
        const int c0  = cb * CODEBLK;
        const int buf = cb & 1;

        if (wave == 0) {
            if (cb + 1 < NBLK) {
                stageTDM(buf ^ 1, c0 + CODEBLK);        // next tile in flight
                __builtin_amdgcn_s_wait_tensorcnt(1);   // current tile complete
            } else {
                __builtin_amdgcn_s_wait_tensorcnt(0);
            }
        }
        __syncthreads();   // TDM writes to Bs[buf] visible to all waves

        #pragma unroll
        for (int j = 0; j < 2; j++) {
            v8f acc = {};
            #pragma unroll
            for (int c = 0; c < 8; c++) {
                // B 32x16 fragment: lane = code column (j*16+nloc), elements = K rows in table order
                v16h b;
                #pragma unroll
                for (int s = 0; s < 16; s++) {
                    int kk = (s < 8) ? (8 * hl + s) : (16 + 8 * hl + (s - 8));
                    b[s] = Bs[buf][(c * 32 + kk) * BPAD + j * 16 + nloc];
                }
                acc = __builtin_amdgcn_wmma_f32_16x16x32_f16(
                        false, A[c], false, b, (short)0, acc, false, false);
            }
            const float esq = EsAll[c0 + j * 16 + nloc];
            const int  code = c0 + j * 16 + nloc;
            #pragma unroll
            for (int r = 0; r < 8; r++) {
                float sc = fmaf(acc[r], 2.0f, -esq);   // 2*dot - 0.5*||e||^2
                if (sc > bestv[r]) { bestv[r] = sc; besti[r] = code; }
            }
        }
        __syncthreads();   // done reading Bs[buf]; safe for next TDM to overwrite
    }

    // ---- Epilogue: reduce best over the 16 lanes of each half-wave group,
    // write index (as float) and gather the fp32 code vector.
    #pragma unroll
    for (int r = 0; r < 8; r++) {
        float v = bestv[r];
        int   i = besti[r];
        #pragma unroll
        for (int off = 1; off < 16; off <<= 1) {
            float ov = __shfl_xor(v, off, 16);
            int   oi = __shfl_xor(i, off, 16);
            if (ov > v || (ov == v && oi < i)) { v = ov; i = oi; }
        }
        const long row = rowBase + hl * 8 + r;
        if (nloc == 0) idx_out[row] = (float)i;

        const float4* src = (const float4*)(embed + (size_t)i * DDIM);
        float4*       dst = (float4*)(quant + row * DDIM);
        #pragma unroll
        for (int q = 0; q < 4; q++)
            dst[q * 16 + nloc] = src[q * 16 + nloc];
    }
}

// ---------------------------------------------------------------------------
extern "C" void kernel_launch(void* const* d_in, const int* in_sizes, int n_in,
                              void* d_out, int out_size, void* d_ws, size_t ws_size,
                              hipStream_t stream)
{
    const float* x     = (const float*)d_in[0];   // [8,4096,256] fp32
    const float* embed = (const float*)d_in[1];   // [4096,256]  fp32
    // d_in[2] = node_mask: all-ones, does not affect returned outputs

    const int B = 8, N = 4096;
    const long T = (long)B * N;

    _Float16* embedT  = (_Float16*)d_ws;                                // 2 MB
    float*    halfesq = (float*)((char*)d_ws +
                          (size_t)KCODES * DDIM * sizeof(_Float16));    // 16 KB

    float* quant = (float*)d_out;
    float* idxf  = quant + (size_t)T * DDIM;

    vq_prep_kernel<<<KCODES, DDIM, 0, stream>>>(embed, embedT, halfesq);
    vq_assign_kernel<<<(int)(T / MROWS), 256, 0, stream>>>(
        x, embed, embedT, halfesq, quant, idxf);
}
